// MidLevel_31756988186949
// MI455X (gfx1250) — matrix operand order repaired and checked by reading, not compiled, not run on hardware
//
#include <hip/hip_runtime.h>

// ---------------------------------------------------------------------------
// CDNA5 / gfx1250: channel-GEMM convolutions on v_wmma_f32_16x16x32_f16.
// Activations channels-last f16 (voxel record = 32 halfs = 64B, matching the
// 16x16x32 A layout). Weights pre-packed to B-register layout and staged in
// LDS per workgroup (shared by its 8 waves). ZT=2 z-blocking in the convs:
// one wave produces two z output planes, reusing each A plane for 2 z-taps.
// ---------------------------------------------------------------------------

typedef __attribute__((ext_vector_type(16))) _Float16 v16h;
typedef __attribute__((ext_vector_type(8)))  _Float16 v8h;
typedef __attribute__((ext_vector_type(8)))  float    v8f;

#define NBLK 64
#define UPS  26
#define C0S  24
#define C1S  22
#define C2S  20
#define SUBV 12
#define DVOL 74
#define OUT_PRED_OFF 28311552ull        // 512*32*12^3
#define WB_HALFS   (27 * 2 * 32 * 16)   // 27648 halfs
#define WB_BYTES   (WB_HALFS * 2)       // 55296 B of LDS (conv kernels)
#define WBUP_HALFS (64 * 2 * 32 * 16)   // 65536 halfs
#define WBUP_BYTES (WBUP_HALFS * 2)     // 131072 B of LDS (upsample kernel)

// workspace layout (bytes)
static constexpr size_t OFF_WBUP = 0;                 // 131072 B
static constexpr size_t OFF_WB0  = 131072;            // 55296 B
static constexpr size_t OFF_WB1  = 186368;
static constexpr size_t OFF_WB2  = 241664;
static constexpr size_t OFF_W0T  = 296960;            // 3456 B
static constexpr size_t OFF_PCL  = 524288;            // 7,077,888 B
static constexpr size_t OFF_X0   = (8ull  << 20);     // 71,991,296 B
static constexpr size_t OFF_X1   = (81ull << 20);     // 56,623,104 B
static constexpr size_t OFF_X2   = OFF_X0;            // reuse (X0 dead after conv0)

__device__ __forceinline__ v8f wmma16(v16h a, v16h b, v8f c) {
  return __builtin_amdgcn_wmma_f32_16x16x32_f16(false, a, false, b, (short)0, c,
                                                false, false);
}

__device__ __forceinline__ v16h zero_a() {
  v16h r;
#pragma unroll
  for (int i = 0; i < 16; ++i) r[i] = (_Float16)0.f;
  return r;
}

// A-matrix per-lane fragment from a 64B channels-last voxel record:
// lanes 0-15 take K=0..7 & 16..23, lanes 16-31 take K=8..15 & 24..31.
__device__ __forceinline__ v16h load_a(const _Float16* p, int g) {
  const v8h* q = (const v8h*)p;
  v8h lo = q[g];
  v8h hi = q[g + 2];
  v16h r;
#pragma unroll
  for (int i = 0; i < 8; ++i) { r[i] = lo[i]; r[i + 8] = hi[i]; }
  return r;
}

__device__ __forceinline__ v16h load_b_lds(const _Float16* WBs, int t, int h, int lane) {
  return *(const v16h*)(WBs + (((t * 2 + h) * 32 + lane) << 4));
}

template <int HALFS>
__device__ __forceinline__ void stage_wb_lds(const _Float16* __restrict__ WB,
                                             _Float16* __restrict__ WBs) {
  const uint4* src = (const uint4*)WB;
  uint4* dst = (uint4*)WBs;
  for (int i = threadIdx.x; i < HALFS / 8; i += 256) dst[i] = src[i];
  __syncthreads();
}

// ---------------------------------------------------------------------------
// ZT=2 conv core: two z output planes per wave. INS = OUTS+2, OUTS even, so
// every plane/z guard is statically true -> no runtime branching near WMMA.
// A rows are predicated in data (zeroed) for p >= OUTS^2 tail lanes.
// ---------------------------------------------------------------------------
template <int INS, int OUTS>
__device__ __forceinline__ void conv_core2(const _Float16* __restrict__ Xin,
                                           const _Float16* __restrict__ WBs,
                                           int n, int z0, int pt, int lane,
                                           v8f& a00, v8f& a01, v8f& a10, v8f& a11) {
  const int m = lane & 15, g = lane >> 4;
  const int p = pt * 16 + m;
  const int py = p / OUTS, px = p % OUTS;
  const bool okrow = (p < OUTS * OUTS);
  const size_t nbase = (size_t)n * INS * INS * INS;
#pragma unroll
  for (int q = 0; q < 4; ++q) {          // input z planes z0 .. z0+3
    const int zi = z0 + q;
    for (int ty = 0; ty < 3; ++ty) {
#pragma unroll
      for (int tx = 0; tx < 3; ++tx) {
        v16h a;
        if (okrow)
          a = load_a(Xin + (nbase +
                     (size_t)(zi * INS * INS + (py + ty) * INS + px + tx)) * 32, g);
        else
          a = zero_a();
        if (q <= 2) {                    // tap tz = q for zout = z0
          const int t = (q * 3 + ty) * 3 + tx;
          a00 = wmma16(a, load_b_lds(WBs, t, 0, lane), a00);
          a01 = wmma16(a, load_b_lds(WBs, t, 1, lane), a01);
        }
        if (q >= 1) {                    // tap tz = q-1 for zout = z0+1
          const int t = ((q - 1) * 3 + ty) * 3 + tx;
          a10 = wmma16(a, load_b_lds(WBs, t, 0, lane), a10);
          a11 = wmma16(a, load_b_lds(WBs, t, 1, lane), a11);
        }
      }
    }
  }
}

// ---------------------------------------------------------------------------
// K0: pack weights -> f16 B-register layout WB[tap][ocHalf][lane][16]
// ---------------------------------------------------------------------------
__global__ __launch_bounds__(256) void prep_weights_kernel(
    const float* __restrict__ w_up, const float* __restrict__ w_c0,
    const float* __restrict__ w_c1, const float* __restrict__ w_c2,
    _Float16* __restrict__ WBup, _Float16* __restrict__ WB0,
    _Float16* __restrict__ WB1, _Float16* __restrict__ WB2,
    float* __restrict__ W0T) {
  int idx = blockIdx.x * 256 + threadIdx.x;
  if (idx < WBUP_HALFS) {         // upsample (ci,co,4,4,4)
    int j = idx & 15, lane = (idx >> 4) & 31, h = (idx >> 9) & 1, t = idx >> 10;
    int ci = (lane >> 4) * 8 + (j & 7) + (j >> 3) * 16, co = h * 16 + (lane & 15);
    WBup[idx] = (_Float16)w_up[(ci * 32 + co) * 64 + t];
  }
  if (idx < WB_HALFS) {           // conv weights
    int j = idx & 15, lane = (idx >> 4) & 31, h = (idx >> 9) & 1, t = idx >> 10;
    int ic = (lane >> 4) * 8 + (j & 7) + (j >> 3) * 16, oc = h * 16 + (lane & 15);
    WB0[idx] = (_Float16)w_c0[(oc * 33 + ic) * 27 + t];   // (32,33,27)
    WB1[idx] = (_Float16)w_c1[(oc * 32 + ic) * 27 + t];   // (32,32,27)
    WB2[idx] = (_Float16)w_c2[(oc * 32 + ic) * 27 + t];
  }
  if (idx < 27 * 32) {            // conv0 tsdf channel (ic == 32)
    int t = idx / 32, oc = idx % 32;
    W0T[idx] = w_c0[(oc * 33 + 32) * 27 + t];
  }
}

// ---------------------------------------------------------------------------
// K1: prev_blocks (N,32,12^3) f32 -> channels-last f16 Pcl (N,12^3,32)
// ---------------------------------------------------------------------------
__global__ __launch_bounds__(256) void transpose_prev_kernel(
    const float* __restrict__ prev, _Float16* __restrict__ Pcl) {
  int idx = blockIdx.x * 256 + threadIdx.x;
  int c = idx & 31;
  int v = (idx >> 5) % 1728;
  int n = (idx >> 5) / 1728;
  Pcl[idx] = (_Float16)prev[((size_t)(n * 32 + c)) * 1728 + v];
}

// ---------------------------------------------------------------------------
// K2: ConvTranspose3d(32->32,k=4,s=2) via WMMA; per-lane parity predication
// zeroes invalid A rows so EXEC stays all-ones across the WMMAs. B tap set
// (131 KB) staged in LDS once per workgroup.
// ---------------------------------------------------------------------------
__global__ __launch_bounds__(256) void upsample_kernel(
    const _Float16* __restrict__ Pcl, const _Float16* __restrict__ WBup,
    _Float16* __restrict__ X0) {
  extern __shared__ _Float16 WBs[];
  stage_wb_lds<WBUP_HALFS>(WBup, WBs);

  int wid = blockIdx.x * 8 + (threadIdx.x >> 5);
  if (wid >= NBLK * UPS * UPS * 2) return;
  int lane = threadIdx.x & 31;
  int xt = wid & 1; int rest = wid >> 1;
  int oy = rest % UPS; rest /= UPS;
  int oz = rest % UPS; int n = rest / UPS;

  const int m = lane & 15, g = lane >> 4;
  const int ox = xt * 16 + m;
  v8f acc0 = {}, acc1 = {};

  for (int kz = 0; kz < 4; ++kz) {
    int izt = oz - kz; if (izt < 0 || (izt & 1)) continue;
    int iz = izt >> 1; if (iz >= SUBV) continue;
    for (int ky = 0; ky < 4; ++ky) {
      int iyt = oy - ky; if (iyt < 0 || (iyt & 1)) continue;
      int iy = iyt >> 1; if (iy >= SUBV) continue;
#pragma unroll
      for (int kx = 0; kx < 4; ++kx) {
        int ixt = ox - kx;
        bool ok = (ixt >= 0) && !(ixt & 1) && ((ixt >> 1) < SUBV);
        int ix = ixt >> 1;
        v16h a;
        if (ok)
          a = load_a(Pcl + ((size_t)n * 1728 +
                            (size_t)(iz * 144 + iy * 12 + ix)) * 32, g);
        else
          a = zero_a();
        int t = (kz * 4 + ky) * 4 + kx;
        acc0 = wmma16(a, load_b_lds(WBs, t, 0, lane), acc0);
        acc1 = wmma16(a, load_b_lds(WBs, t, 1, lane), acc1);
      }
    }
  }
  int oc = lane & 15;
#pragma unroll
  for (int r = 0; r < 8; ++r) {
    int M = r + 8 * g; int oxr = xt * 16 + M;
    if (oxr < UPS) {
      size_t base = ((size_t)n * 17576 +
                     (size_t)((oz * UPS + oy) * UPS + oxr)) * 32;
      X0[base + oc]      = (_Float16)acc0[r];
      X0[base + oc + 16] = (_Float16)acc1[r];
    }
  }
}

// ---------------------------------------------------------------------------
// K3: conv0 (33ch->32): WMMA over 32 channels + VALU tsdf channel, BN+ReLU
// ---------------------------------------------------------------------------
__global__ __launch_bounds__(256) void conv0_kernel(
    const _Float16* __restrict__ X0, const _Float16* __restrict__ WB0,
    const float* __restrict__ W0T, const float* __restrict__ tsdf,
    const float* __restrict__ g_bn, const float* __restrict__ b_bn,
    _Float16* __restrict__ X1) {
  extern __shared__ _Float16 WBs[];
  stage_wb_lds<WB_HALFS>(WB0, WBs);

  int wid = blockIdx.x * 8 + (threadIdx.x >> 5);
  if (wid >= NBLK * 12 * 36) return;
  int lane = threadIdx.x & 31;
  int pt = wid % 36; int rest = wid / 36;
  int zg = rest % 12; int n = rest / 12;
  int z0 = zg * 2;

  v8f a00 = {}, a01 = {}, a10 = {}, a11 = {};
  conv_core2<UPS, C0S>(X0, WBs, n, z0, pt, lane, a00, a01, a10, a11);

  // tsdf channel (input channel 32), direct gather
  int g = lane >> 4, oc = lane & 15;
  int bz = (n >> 4) * 16, by = ((n >> 2) & 3) * 16, bx = (n & 3) * 16;
  for (int t = 0; t < 27; ++t) {
    int tz = t / 9, ty = (t / 3) % 3, tx = t % 3;
    float w0 = W0T[t * 32 + oc], w1 = W0T[t * 32 + oc + 16];
#pragma unroll
    for (int r = 0; r < 8; ++r) {
      int p_r = pt * 16 + r + 8 * g;
      if (p_r < C0S * C0S) {
        int y_r = p_r / C0S, x_r = p_r % C0S;
        size_t tb = (size_t)(by + y_r + ty) * DVOL + (bx + x_r + tx);
        float tv0 = tsdf[(size_t)(bz + z0 + tz) * (DVOL * DVOL) + tb];
        float tv1 = tsdf[(size_t)(bz + z0 + 1 + tz) * (DVOL * DVOL) + tb];
        a00[r] += tv0 * w0; a01[r] += tv0 * w1;
        a10[r] += tv1 * w0; a11[r] += tv1 * w1;
      }
    }
  }

  float inv = rsqrtf(1.0f + 1e-5f);
  float s0 = g_bn[oc] * inv, o0 = b_bn[oc];
  float s1 = g_bn[oc + 16] * inv, o1 = b_bn[oc + 16];
#pragma unroll
  for (int r = 0; r < 8; ++r) {
    int p_r = pt * 16 + r + 8 * g;
    if (p_r < C0S * C0S) {
      size_t b0i = ((size_t)n * (C0S * C0S * C0S) +
                    (size_t)(z0 * C0S * C0S) + p_r) * 32;
      size_t b1i = b0i + (size_t)(C0S * C0S) * 32;
      float v;
      v = a00[r] * s0 + o0; X1[b0i + oc]      = (_Float16)(v > 0.f ? v : 0.f);
      v = a01[r] * s1 + o1; X1[b0i + oc + 16] = (_Float16)(v > 0.f ? v : 0.f);
      v = a10[r] * s0 + o0; X1[b1i + oc]      = (_Float16)(v > 0.f ? v : 0.f);
      v = a11[r] * s1 + o1; X1[b1i + oc + 16] = (_Float16)(v > 0.f ? v : 0.f);
    }
  }
}

// ---------------------------------------------------------------------------
// K4: conv1 (32->32) WMMA + BN + ReLU
// ---------------------------------------------------------------------------
__global__ __launch_bounds__(256) void conv1_kernel(
    const _Float16* __restrict__ X1, const _Float16* __restrict__ WB1,
    const float* __restrict__ g_bn, const float* __restrict__ b_bn,
    _Float16* __restrict__ X2) {
  extern __shared__ _Float16 WBs[];
  stage_wb_lds<WB_HALFS>(WB1, WBs);

  int wid = blockIdx.x * 8 + (threadIdx.x >> 5);
  if (wid >= NBLK * 11 * 31) return;
  int lane = threadIdx.x & 31;
  int pt = wid % 31; int rest = wid / 31;
  int zg = rest % 11; int n = rest / 11;
  int z0 = zg * 2;

  v8f a00 = {}, a01 = {}, a10 = {}, a11 = {};
  conv_core2<C0S, C1S>(X1, WBs, n, z0, pt, lane, a00, a01, a10, a11);

  int g = lane >> 4, oc = lane & 15;
  float inv = rsqrtf(1.0f + 1e-5f);
  float s0 = g_bn[oc] * inv, o0 = b_bn[oc];
  float s1 = g_bn[oc + 16] * inv, o1 = b_bn[oc + 16];
#pragma unroll
  for (int r = 0; r < 8; ++r) {
    int p_r = pt * 16 + r + 8 * g;
    if (p_r < C1S * C1S) {
      size_t b0i = ((size_t)n * (C1S * C1S * C1S) +
                    (size_t)(z0 * C1S * C1S) + p_r) * 32;
      size_t b1i = b0i + (size_t)(C1S * C1S) * 32;
      float v;
      v = a00[r] * s0 + o0; X2[b0i + oc]      = (_Float16)(v > 0.f ? v : 0.f);
      v = a01[r] * s1 + o1; X2[b0i + oc + 16] = (_Float16)(v > 0.f ? v : 0.f);
      v = a10[r] * s0 + o0; X2[b1i + oc]      = (_Float16)(v > 0.f ? v : 0.f);
      v = a11[r] * s1 + o1; X2[b1i + oc + 16] = (_Float16)(v > 0.f ? v : 0.f);
    }
  }
}

// ---------------------------------------------------------------------------
// K5: conv2 (32->32) WMMA + BN + ReLU + fused split_tree scatter (f32 out)
// ---------------------------------------------------------------------------
__global__ __launch_bounds__(256) void conv2_kernel(
    const _Float16* __restrict__ X2, const _Float16* __restrict__ WB2,
    const float* __restrict__ g_bn, const float* __restrict__ b_bn,
    float* __restrict__ out) {
  extern __shared__ _Float16 WBs[];
  stage_wb_lds<WB_HALFS>(WB2, WBs);

  int wid = blockIdx.x * 8 + (threadIdx.x >> 5);
  if (wid >= NBLK * 10 * 25) return;
  int lane = threadIdx.x & 31;
  int pt = wid % 25; int rest = wid / 25;
  int zg = rest % 10; int n = rest / 10;
  int z0 = zg * 2;

  v8f a00 = {}, a01 = {}, a10 = {}, a11 = {};
  conv_core2<C1S, C2S>(X2, WBs, n, z0, pt, lane, a00, a01, a10, a11);

  int g = lane >> 4, oc = lane & 15;
  float inv = rsqrtf(1.0f + 1e-5f);
  float s0 = g_bn[oc] * inv, o0 = b_bn[oc];
  float s1 = g_bn[oc + 16] * inv, o1 = b_bn[oc + 16];
#pragma unroll
  for (int r = 0; r < 8; ++r) {
    int p_r = pt * 16 + r + 8 * g;
    if (p_r >= C2S * C2S) continue;
    int y_r = p_r / C2S, x_r = p_r % C2S;
#pragma unroll
    for (int dz = 0; dz < 2; ++dz) {
      int zout = z0 + dz;
      float v0 = (dz ? a10[r] : a00[r]) * s0 + o0; v0 = v0 > 0.f ? v0 : 0.f;
      float v1 = (dz ? a11[r] : a01[r]) * s1 + o1; v1 = v1 > 0.f ? v1 : 0.f;
      for (int a = 0; a < 2; ++a) {
        int za = zout - a * 8; if (za < 0 || za >= SUBV) continue;
        for (int b = 0; b < 2; ++b) {
          int yb = y_r - b * 8; if (yb < 0 || yb >= SUBV) continue;
          for (int c = 0; c < 2; ++c) {
            int xc = x_r - c * 8; if (xc < 0 || xc >= SUBV) continue;
            size_t o = ((size_t)((n * 8 + a * 4 + b * 2 + c) * 32) + oc) * 1728 +
                       (size_t)(za * 144 + yb * 12 + xc);
            out[o]             = v0;
            out[o + 16 * 1728] = v1;
          }
        }
      }
    }
  }
}

// ---------------------------------------------------------------------------
// K6: prediction path (conv->bn->relu->pool x2, linear) — ~1% of FLOPs, VALU
// ---------------------------------------------------------------------------
__global__ __launch_bounds__(256) void pred_kernel(
    const float* __restrict__ prev,
    const float* __restrict__ w_p0, const float* __restrict__ g_p0,
    const float* __restrict__ b_p0, const float* __restrict__ w_p1,
    const float* __restrict__ g_p1, const float* __restrict__ b_p1,
    const float* __restrict__ w_lin, const float* __restrict__ b_lin,
    float* __restrict__ outp) {
  __shared__ float pool0[2000];
  __shared__ float h1[216];
  __shared__ float pool1[8];
  int n = blockIdx.x, tid = threadIdx.x;
  float inv = rsqrtf(1.0f + 1e-5f);

  for (int p = tid; p < 2000; p += 256) {
    int c = p / 125, r = p % 125;
    int pz = r / 25, py = (r / 5) % 5, px = r % 5;
    float mx = -3.4e38f;
    for (int dz = 0; dz < 2; ++dz)
      for (int dy = 0; dy < 2; ++dy)
        for (int dx = 0; dx < 2; ++dx) {
          int cz = 2 * pz + dz, cy = 2 * py + dy, cx = 2 * px + dx;
          float s = 0.f;
          for (int ci = 0; ci < 32; ++ci) {
            const float* pv = prev + ((size_t)(n * 32 + ci)) * 1728;
            const float* wv = w_p0 + ((size_t)(c * 32 + ci)) * 27;
            for (int kz = 0; kz < 3; ++kz)
              for (int ky = 0; ky < 3; ++ky)
                for (int kx = 0; kx < 3; ++kx)
                  s += pv[(cz + kz) * 144 + (cy + ky) * 12 + (cx + kx)] *
                       wv[(kz * 3 + ky) * 3 + kx];
          }
          s = s * g_p0[c] * inv + b_p0[c];
          s = s > 0.f ? s : 0.f;
          mx = s > mx ? s : mx;
        }
    pool0[p] = mx;
  }
  __syncthreads();
  if (tid < 216) {
    int c = tid / 27, r = tid % 27;
    int cz = r / 9, cy = (r / 3) % 3, cx = r % 3;
    float s = 0.f;
    for (int ci = 0; ci < 16; ++ci) {
      const float* wv = w_p1 + ((size_t)(c * 16 + ci)) * 27;
      for (int kz = 0; kz < 3; ++kz)
        for (int ky = 0; ky < 3; ++ky)
          for (int kx = 0; kx < 3; ++kx)
            s += pool0[ci * 125 + (cz + kz) * 25 + (cy + ky) * 5 + (cx + kx)] *
                 wv[(kz * 3 + ky) * 3 + kx];
    }
    s = s * g_p1[c] * inv + b_p1[c];
    h1[tid] = s > 0.f ? s : 0.f;
  }
  __syncthreads();
  if (tid < 8) {
    float mx = -3.4e38f;
    for (int dz = 0; dz < 2; ++dz)
      for (int dy = 0; dy < 2; ++dy)
        for (int dx = 0; dx < 2; ++dx) {
          float v = h1[tid * 27 + dz * 9 + dy * 3 + dx];
          mx = v > mx ? v : mx;
        }
    pool1[tid] = mx;
  }
  __syncthreads();
  if (tid < 3) {
    float s = b_lin[tid];
    for (int c = 0; c < 8; ++c) s += w_lin[tid * 8 + c] * pool1[c];
    outp[n * 3 + tid] = s;
  }
}

// ---------------------------------------------------------------------------
extern "C" void kernel_launch(void* const* d_in, const int* in_sizes, int n_in,
                              void* d_out, int out_size, void* d_ws, size_t ws_size,
                              hipStream_t stream) {
  const float* prev  = (const float*)d_in[0];
  const float* tsdf  = (const float*)d_in[1];
  const float* w_up  = (const float*)d_in[2];
  const float* w_c0  = (const float*)d_in[3];
  const float* g_c0  = (const float*)d_in[4];
  const float* b_c0  = (const float*)d_in[5];
  const float* w_c1  = (const float*)d_in[6];
  const float* g_c1  = (const float*)d_in[7];
  const float* b_c1  = (const float*)d_in[8];
  const float* w_c2  = (const float*)d_in[9];
  const float* g_c2  = (const float*)d_in[10];
  const float* b_c2  = (const float*)d_in[11];
  const float* w_p0  = (const float*)d_in[12];
  const float* g_p0  = (const float*)d_in[13];
  const float* b_p0  = (const float*)d_in[14];
  const float* w_p1  = (const float*)d_in[15];
  const float* g_p1  = (const float*)d_in[16];
  const float* b_p1  = (const float*)d_in[17];
  const float* w_lin = (const float*)d_in[18];
  const float* b_lin = (const float*)d_in[19];
  float* out = (float*)d_out;
  char* ws = (char*)d_ws;

  _Float16* WBup = (_Float16*)(ws + OFF_WBUP);
  _Float16* WB0  = (_Float16*)(ws + OFF_WB0);
  _Float16* WB1  = (_Float16*)(ws + OFF_WB1);
  _Float16* WB2  = (_Float16*)(ws + OFF_WB2);
  float*    W0T  = (float*)   (ws + OFF_W0T);
  _Float16* Pcl  = (_Float16*)(ws + OFF_PCL);
  _Float16* X0   = (_Float16*)(ws + OFF_X0);
  _Float16* X1   = (_Float16*)(ws + OFF_X1);
  _Float16* X2   = (_Float16*)(ws + OFF_X2);

  prep_weights_kernel<<<256, 256, 0, stream>>>(w_up, w_c0, w_c1, w_c2,
                                               WBup, WB0, WB1, WB2, W0T);
  transpose_prev_kernel<<<(NBLK * 32 * 1728) / 256, 256, 0, stream>>>(prev, Pcl);
  upsample_kernel<<<(NBLK * UPS * UPS * 2) / 8, 256, WBUP_BYTES, stream>>>(Pcl, WBup, X0);
  conv0_kernel<<<(NBLK * 12 * 36) / 8, 256, WB_BYTES, stream>>>(X0, WB0, W0T, tsdf,
                                                                g_c0, b_c0, X1);
  conv1_kernel<<<(NBLK * 11 * 31) / 8, 256, WB_BYTES, stream>>>(X1, WB1, g_c1, b_c1, X2);
  conv2_kernel<<<(NBLK * 10 * 25) / 8, 256, WB_BYTES, stream>>>(X2, WB2, g_c2, b_c2, out);
  pred_kernel<<<NBLK, 256, 0, stream>>>(prev, w_p0, g_p0, b_p0, w_p1, g_p1, b_p1,
                                        w_lin, b_lin, out + OUT_PRED_OFF);
}
